// ChanelSpace_Attn_463856468505
// MI455X (gfx1250) — matrix operand order repaired and checked by reading, compiled, not run on hardware
//
#include <hip/hip_runtime.h>

// MI455X / gfx1250, wave32. All large GEMMs use v_wmma_f32_16x16x32_f16.
// Fused flash-attention (online softmax, base-2 exp with log2e folded into q)
// avoids materializing the [8,4096,1024] energy/attn tensors.
// K-loops use copy-free ping-pong double buffering: two register sets,
// refilled directly by loads (no buffer-rotation movs).

typedef __attribute__((ext_vector_type(16))) _Float16 v16h;
typedef __attribute__((ext_vector_type(8)))  float    v8f;

#define DEVI __device__ __forceinline__

constexpr int B_ = 8, C_ = 512, Wd = 64, Hd = 64;
constexpr int N_ = Wd * Hd;        // 4096
constexpr int M_ = N_ / 4;         // 1024 (after 2x2 maxpool)
constexpr int CQ = C_ / 8;         // 64
constexpr int CV = C_ / 2;         // 256
constexpr int OC = CQ + CQ + CV;   // 384 (q|k|v output channels)
constexpr float LOG2E = 1.4426950408889634f;

// Native v_exp_f32 (exp2).
DEVI float fast_exp2(float x) { return __builtin_amdgcn_exp2f(x); }

// Load a v16h WMMA fragment from two 16B-aligned 16-byte chunks.
DEVI v16h ldpair(const _Float16* p0, const _Float16* p1) {
  union { uint4 u[2]; v16h v; } t;
  t.u[0] = *reinterpret_cast<const uint4*>(p0);
  t.u[1] = *reinterpret_cast<const uint4*>(p1);
  return t.v;
}

DEVI v8f wmma_f16(v16h a, v16h b, v8f c) {
  return __builtin_amdgcn_wmma_f32_16x16x32_f16(false, a, false, b, (short)0, c,
                                                false, false);
}

// ---------------------------------------------------------------- weights
__global__ void pack_weights(const float* wq, const float* wk, const float* wv,
                             const float* wo, const float* bq, const float* bk,
                             const float* bv, _Float16* wqkv, _Float16* woh,
                             float* biasqkv) {
  int i = blockIdx.x * 256 + threadIdx.x;
  const int n_qkv = OC * C_;        // 196608
  const int n_wo  = C_ * CV;        // 131072
  if (i < n_qkv) {
    int row = i / C_, col = i - row * C_;
    float v;
    if (row < CQ)            v = wq[row * C_ + col];
    else if (row < 2 * CQ)   v = wk[(row - CQ) * C_ + col];
    else                     v = wv[(row - 2 * CQ) * C_ + col];
    wqkv[i] = (_Float16)v;
  } else if (i < n_qkv + n_wo) {
    int j = i - n_qkv;
    woh[j] = (_Float16)wo[j];
  } else if (i < n_qkv + n_wo + OC) {
    int r = i - n_qkv - n_wo;
    biasqkv[r] = (r < CQ) ? bq[r] : (r < 2 * CQ ? bk[r - CQ] : bv[r - 2 * CQ]);
  }
}

// ---------------------------------------------------- SE gate (tiny GEMMs)
__global__ void channel_mean(const float* x, float* y0) {
  int bc = blockIdx.x;                       // b*C + c
  const float* p = x + (size_t)bc * N_;
  float s = 0.f;
  for (int i = threadIdx.x; i < N_; i += 256) s += p[i];
  __shared__ float red[256];
  red[threadIdx.x] = s;
  __syncthreads();
  for (int st = 128; st > 0; st >>= 1) {
    if (threadIdx.x < st) red[threadIdx.x] += red[threadIdx.x + st];
    __syncthreads();
  }
  if (threadIdx.x == 0) y0[bc] = red[0] * (1.0f / N_);
}

__global__ void se_fc1(const float* y0, const float* fc1, float* h) {
  int b = blockIdx.x, j = threadIdx.x;       // 256 threads
  const float* yb = y0 + b * C_;
  const float* w  = fc1 + (size_t)j * C_;
  float s = 0.f;
  for (int c = 0; c < C_; ++c) s += yb[c] * w[c];
  h[b * CV + j] = s > 0.f ? s : 0.f;
}

__global__ void se_fc2(const float* h, const float* fc2, float* y) {
  int b = blockIdx.x, o = threadIdx.x;       // 512 threads
  const float* hb = h + b * CV;
  const float* w  = fc2 + (size_t)o * CV;
  float s = 0.f;
  for (int j = 0; j < CV; ++j) s += hb[j] * w[j];
  // sigmoid(s) = 1/(1+exp(-s)) = 1/(1+exp2(-s*log2e))
  y[b * C_ + o] = 1.f / (1.f + fast_exp2(-s * LOG2E));
}

// ------------------------------------------- x [B,C,N] f32 -> xT [B,N,C] f16
__global__ void transpose_x(const float* x, _Float16* xT) {
  __shared__ __align__(16) _Float16 tile[32][65];
  int b = blockIdx.z, c0 = blockIdx.y * 32, n0 = blockIdx.x * 64;
  int tid = threadIdx.x;
  int jr = tid & 63, ir = tid >> 6;          // 4 c-rows per pass (coalesced n)
#pragma unroll
  for (int p = 0; p < 8; ++p) {
    int c = ir + 4 * p;
    tile[c][jr] = (_Float16)x[((size_t)b * C_ + c0 + c) * N_ + n0 + jr];
  }
  __syncthreads();
  int cw = tid & 31, nw = tid >> 5;          // 8 n-rows per pass (coalesced c)
#pragma unroll
  for (int p = 0; p < 8; ++p) {
    int n = nw + 8 * p;
    xT[((size_t)b * N_ + n0 + n) * C_ + c0 + cw] = tile[cw][n];
  }
}

// --------------------------- QKV conv1x1 as WMMA GEMM: [n,512] x [512,oc]
// block: 256 thr = 8 waves; wave = 16 n-rows x 64 oc; grid (OC/64, N/128, B)
// Copy-free ping-pong K-loop: 2 K-steps (64) per iteration, 8 iterations.
__global__ void __launch_bounds__(256) qkv_gemm(
    const _Float16* xT, const _Float16* wqkv, const float* biasqkv,
    _Float16* qT, _Float16* kFT, _Float16* vFT) {
  int lane = threadIdx.x & 31, wave = threadIdx.x >> 5;
  int kh = lane >> 4, ln = lane & 15;
  int b   = blockIdx.z;
  int oc0 = blockIdx.x * 64;
  int n0  = blockIdx.y * 128 + wave * 16;
  const _Float16* abase = xT + ((size_t)b * N_ + n0 + ln) * C_;
  const _Float16* bbase[4];
#pragma unroll
  for (int t = 0; t < 4; ++t)
    bbase[t] = wqkv + (size_t)(oc0 + 16 * t + ln) * C_ + 16 * kh;

  v8f z = {0.f, 0.f, 0.f, 0.f, 0.f, 0.f, 0.f, 0.f};
  v8f acc[4] = {z, z, z, z};
  v16h aC = ldpair(abase + 16 * kh, abase + 16 * kh + 8);
  v16h bC[4];
#pragma unroll
  for (int t = 0; t < 4; ++t) bC[t] = ldpair(bbase[t], bbase[t] + 8);

  for (int k0 = 0; k0 < C_; k0 += 64) {
    int k1 = k0 + 32;                         // always < C_
    v16h aN = ldpair(abase + k1 + 16 * kh, abase + k1 + 16 * kh + 8);
    v16h bN[4];
#pragma unroll
    for (int t = 0; t < 4; ++t)
      bN[t] = ldpair(bbase[t] + k1, bbase[t] + k1 + 8);
#pragma unroll
    for (int t = 0; t < 4; ++t) acc[t] = wmma_f16(aC, bC[t], acc[t]);
    if (k0 + 64 < C_) {                       // refill C-set in place
      int k2 = k0 + 64;
      aC = ldpair(abase + k2 + 16 * kh, abase + k2 + 16 * kh + 8);
#pragma unroll
      for (int t = 0; t < 4; ++t)
        bC[t] = ldpair(bbase[t] + k2, bbase[t] + k2 + 8);
    }
#pragma unroll
    for (int t = 0; t < 4; ++t) acc[t] = wmma_f16(aN, bN[t], acc[t]);
  }

  _Float16* dst;
  int ocAdj, width;
  float sc;  // fold log2(e) into q so attention softmax can use exp2
  if (oc0 < CQ)          { dst = qT;  ocAdj = 0;      width = CQ; sc = LOG2E; }
  else if (oc0 < 2 * CQ) { dst = kFT; ocAdj = CQ;     width = CQ; sc = 1.f; }
  else                   { dst = vFT; ocAdj = 2 * CQ; width = CV; sc = 1.f; }
#pragma unroll
  for (int t = 0; t < 4; ++t) {
    int oc = oc0 + 16 * t + ln;
    float bias = biasqkv[oc];
#pragma unroll
    for (int r = 0; r < 8; ++r) {
      int n = n0 + r + 8 * kh;
      dst[((size_t)b * N_ + n) * width + (oc - ocAdj)] =
          (_Float16)((acc[t][r] + bias) * sc);
    }
  }
}

// ------------------- 2x2 maxpool: kFT[B,N,64]->kT[B,M,64]; vFT->vP[B,256,M]
__global__ void pool_kv(const _Float16* kFT, const _Float16* vFT,
                        _Float16* kT, _Float16* vP) {
  int b = blockIdx.y, m = blockIdx.x;
  int wi = m >> 5, hi = m & 31;
  int n00 = (2 * wi) * Hd + 2 * hi;
  int c = threadIdx.x;                       // 256 threads
  const _Float16* vb = vFT + (size_t)b * N_ * CV;
  float v0 = (float)vb[(size_t)(n00)          * CV + c];
  float v1 = (float)vb[(size_t)(n00 + 1)      * CV + c];
  float v2 = (float)vb[(size_t)(n00 + Hd)     * CV + c];
  float v3 = (float)vb[(size_t)(n00 + Hd + 1) * CV + c];
  vP[((size_t)b * CV + c) * M_ + m] = (_Float16)fmaxf(fmaxf(v0, v1), fmaxf(v2, v3));
  if (c < CQ) {
    const _Float16* kb = kFT + (size_t)b * N_ * CQ;
    float k0 = (float)kb[(size_t)(n00)          * CQ + c];
    float k1 = (float)kb[(size_t)(n00 + 1)      * CQ + c];
    float k2 = (float)kb[(size_t)(n00 + Hd)     * CQ + c];
    float k3 = (float)kb[(size_t)(n00 + Hd + 1) * CQ + c];
    kT[((size_t)b * M_ + m) * CQ + c] = (_Float16)fmaxf(fmaxf(k0, k1), fmaxf(k2, k3));
  }
}

// ---------------------------------------- fused attention (flash-style)
// block: 256 thr = 8 waves; wave pair shares 16 n-rows, each wave 128 of 256
// value channels. Online softmax (base-2; q pre-scaled by log2e) over M=1024
// keys in steps of 32. v-fragments are loaded before the stat chain and the
// LDS P round-trip so global loads overlap VALU/trans work. Grid (N/64, B).
__global__ void __launch_bounds__(256) attn_kernel(
    const _Float16* qT, const _Float16* kT, const _Float16* vP,
    _Float16* outAT) {
  __shared__ __align__(16) _Float16 pbuf[8][16][32];  // per-wave P tile
  int lane = threadIdx.x & 31, wave = threadIdx.x >> 5;
  int kh = lane >> 4, ln = lane & 15;
  int b     = blockIdx.y;
  int n0    = blockIdx.x * 64 + (wave >> 1) * 16;
  int chalf = (wave & 1) * 128;
  const _Float16* qrow = qT + ((size_t)b * N_ + n0 + ln) * CQ;
  v16h aq0 = ldpair(qrow +      16 * kh, qrow +  8 + 16 * kh);   // K 0..31
  v16h aq1 = ldpair(qrow + 32 + 16 * kh, qrow + 40 + 16 * kh);   // K 32..63
  v8f z = {0.f, 0.f, 0.f, 0.f, 0.f, 0.f, 0.f, 0.f};
  v8f acc[8] = {z, z, z, z, z, z, z, z};
  float mrun[8], lrun[8];
#pragma unroll
  for (int r = 0; r < 8; ++r) { mrun[r] = -3.0e38f; lrun[r] = 0.f; }

  for (int m0 = 0; m0 < M_; m0 += 32) {
    const _Float16* kr0 = kT + ((size_t)b * M_ + m0 + ln) * CQ;
    const _Float16* kr1 = kr0 + 16 * CQ;
    v16h kf0 = ldpair(kr0 + 16 * kh, kr0 + 16 * kh + 8);
    v16h kf1 = ldpair(kr0 + 32 + 16 * kh, kr0 + 40 + 16 * kh);
    v16h kf2 = ldpair(kr1 + 16 * kh, kr1 + 16 * kh + 8);
    v16h kf3 = ldpair(kr1 + 32 + 16 * kh, kr1 + 40 + 16 * kh);
    if (m0 + 32 < M_) {                     // prefetch next key tile
      __builtin_prefetch(kr0 + 32 * CQ, 0, 3);
      __builtin_prefetch(kr1 + 32 * CQ, 0, 3);
    }
    v8f S0 = wmma_f16(aq0, kf0, z);
    S0     = wmma_f16(aq1, kf1, S0);
    v8f S1 = wmma_f16(aq0, kf2, z);
    S1     = wmma_f16(aq1, kf3, S1);

    // Issue all value-fragment loads now; they complete while the softmax
    // stat chain (shuffles + v_exp) and LDS transpose run.
    v16h vf[8];
#pragma unroll
    for (int t = 0; t < 8; ++t) {
      int c = chalf + 16 * t + ln;
      const _Float16* vb = vP + ((size_t)b * CV + c) * M_ + m0 + 16 * kh;
      vf[t] = ldpair(vb, vb + 8);
    }

    float scale[8];
#pragma unroll
    for (int r = 0; r < 8; ++r) {
      float v = fmaxf(S0[r], S1[r]);                 // row = r + 8*kh
      v = fmaxf(v, __shfl_xor(v, 1, 32));
      v = fmaxf(v, __shfl_xor(v, 2, 32));
      v = fmaxf(v, __shfl_xor(v, 4, 32));
      v = fmaxf(v, __shfl_xor(v, 8, 32));
      float mn = fmaxf(mrun[r], v);
      scale[r] = fast_exp2(mrun[r] - mn);
      mrun[r]  = mn;
      float p0 = fast_exp2(S0[r] - mn), p1 = fast_exp2(S1[r] - mn);
      S0[r] = p0; S1[r] = p1;
      float s = p0 + p1;
      s += __shfl_xor(s, 1, 32);
      s += __shfl_xor(s, 2, 32);
      s += __shfl_xor(s, 4, 32);
      s += __shfl_xor(s, 8, 32);
      lrun[r] = lrun[r] * scale[r] + s;
    }
    // C-layout -> A-layout transpose of P through per-wave LDS
#pragma unroll
    for (int r = 0; r < 8; ++r) {
      pbuf[wave][r + 8 * kh][ln]      = (_Float16)S0[r];
      pbuf[wave][r + 8 * kh][ln + 16] = (_Float16)S1[r];
    }
    asm volatile("s_wait_dscnt 0" ::: "memory");
    const _Float16* prow = &pbuf[wave][ln][0];
    v16h ap = ldpair(prow + 16 * kh, prow + 16 * kh + 8);
#pragma unroll
    for (int t = 0; t < 8; ++t)
#pragma unroll
      for (int r = 0; r < 8; ++r) acc[t][r] *= scale[r];
#pragma unroll
    for (int t = 0; t < 8; ++t) acc[t] = wmma_f16(ap, vf[t], acc[t]);
  }
#pragma unroll
  for (int t = 0; t < 8; ++t) {
    int c = chalf + 16 * t + ln;
#pragma unroll
    for (int r = 0; r < 8; ++r) {
      int n = n0 + r + 8 * kh;
      outAT[((size_t)b * N_ + n) * CV + c] = (_Float16)(acc[t][r] / lrun[r]);
    }
  }
}

// ---------------- final conv1x1 (wo) + gamma*out + x*y epilogue, fp32 out
// wave = 32 o x 32 n; block = 64 o x 128 n; grid (N/128, C/64, B)
// Copy-free ping-pong K-loop: 2 K-steps (64) per iteration, 4 iterations.
__global__ void __launch_bounds__(256) final_gemm(
    const _Float16* outAT, const _Float16* woh, const float* bo,
    const float* x, const float* yg, const float* gamma, float* out) {
  int lane = threadIdx.x & 31, wave = threadIdx.x >> 5;
  int kh = lane >> 4, ln = lane & 15;
  int b  = blockIdx.z;
  int o0 = blockIdx.y * 64 + (wave >> 2) * 32;
  int n0 = blockIdx.x * 128 + (wave & 3) * 32;
  v8f z = {0.f, 0.f, 0.f, 0.f, 0.f, 0.f, 0.f, 0.f};
  v8f acc[2][2] = {{z, z}, {z, z}};
  const _Float16* aR0 = woh + (size_t)(o0 + ln) * CV;
  const _Float16* aR1 = aR0 + 16 * CV;
  const _Float16* bR0 = outAT + ((size_t)b * N_ + n0 + ln) * CV;
  const _Float16* bR1 = bR0 + 16 * CV;

  v16h a0C = ldpair(aR0 + 16 * kh, aR0 + 16 * kh + 8);
  v16h a1C = ldpair(aR1 + 16 * kh, aR1 + 16 * kh + 8);
  v16h b0C = ldpair(bR0 + 16 * kh, bR0 + 16 * kh + 8);
  v16h b1C = ldpair(bR1 + 16 * kh, bR1 + 16 * kh + 8);

  for (int k0 = 0; k0 < CV; k0 += 64) {
    int k1 = k0 + 32;                         // always < CV
    v16h a0N = ldpair(aR0 + k1 + 16 * kh, aR0 + k1 + 16 * kh + 8);
    v16h a1N = ldpair(aR1 + k1 + 16 * kh, aR1 + k1 + 16 * kh + 8);
    v16h b0N = ldpair(bR0 + k1 + 16 * kh, bR0 + k1 + 16 * kh + 8);
    v16h b1N = ldpair(bR1 + k1 + 16 * kh, bR1 + k1 + 16 * kh + 8);
    acc[0][0] = wmma_f16(a0C, b0C, acc[0][0]);
    acc[0][1] = wmma_f16(a0C, b1C, acc[0][1]);
    acc[1][0] = wmma_f16(a1C, b0C, acc[1][0]);
    acc[1][1] = wmma_f16(a1C, b1C, acc[1][1]);
    if (k0 + 64 < CV) {                       // refill C-set in place
      int k2 = k0 + 64;
      a0C = ldpair(aR0 + k2 + 16 * kh, aR0 + k2 + 16 * kh + 8);
      a1C = ldpair(aR1 + k2 + 16 * kh, aR1 + k2 + 16 * kh + 8);
      b0C = ldpair(bR0 + k2 + 16 * kh, bR0 + k2 + 16 * kh + 8);
      b1C = ldpair(bR1 + k2 + 16 * kh, bR1 + k2 + 16 * kh + 8);
    }
    acc[0][0] = wmma_f16(a0N, b0N, acc[0][0]);
    acc[0][1] = wmma_f16(a0N, b1N, acc[0][1]);
    acc[1][0] = wmma_f16(a1N, b0N, acc[1][0]);
    acc[1][1] = wmma_f16(a1N, b1N, acc[1][1]);
  }

  float g = gamma[0];
#pragma unroll
  for (int oi = 0; oi < 2; ++oi)
#pragma unroll
    for (int ni = 0; ni < 2; ++ni)
#pragma unroll
      for (int r = 0; r < 8; ++r) {
        int o = o0 + 16 * oi + r + 8 * kh;
        int n = n0 + 16 * ni + ln;
        size_t idx = ((size_t)b * C_ + o) * N_ + n;
        out[idx] = g * (acc[oi][ni][r] + bo[o]) + x[idx] * yg[b * C_ + o];
      }
}

// ---------------------------------------------------------------- launcher
extern "C" void kernel_launch(void* const* d_in, const int* in_sizes, int n_in,
                              void* d_out, int out_size, void* d_ws,
                              size_t ws_size, hipStream_t stream) {
  const float* x    = (const float*)d_in[0];
  const float* wq   = (const float*)d_in[1];
  const float* bq   = (const float*)d_in[2];
  const float* wk   = (const float*)d_in[3];
  const float* bk   = (const float*)d_in[4];
  const float* wv   = (const float*)d_in[5];
  const float* bv   = (const float*)d_in[6];
  const float* wo   = (const float*)d_in[7];
  const float* bo   = (const float*)d_in[8];
  const float* fc1  = (const float*)d_in[9];
  const float* fc2  = (const float*)d_in[10];
  const float* gamma = (const float*)d_in[11];
  float* out = (float*)d_out;

  char* ws = (char*)d_ws;
  size_t off = 0;
  auto alloc = [&](size_t bytes) {
    size_t o = off;
    off += (bytes + 255) & ~(size_t)255;
    return o;
  };
  _Float16* wqkv_h = (_Float16*)(ws + alloc((size_t)OC * C_ * 2));
  _Float16* wo_h   = (_Float16*)(ws + alloc((size_t)C_ * CV * 2));
  float*    biasq  = (float*)   (ws + alloc((size_t)OC * 4));
  float*    y0     = (float*)   (ws + alloc((size_t)B_ * C_ * 4));
  float*    hbuf   = (float*)   (ws + alloc((size_t)B_ * CV * 4));
  float*    ybuf   = (float*)   (ws + alloc((size_t)B_ * C_ * 4));
  _Float16* qT     = (_Float16*)(ws + alloc((size_t)B_ * N_ * CQ * 2));
  _Float16* kFT    = (_Float16*)(ws + alloc((size_t)B_ * N_ * CQ * 2));
  _Float16* vFT    = (_Float16*)(ws + alloc((size_t)B_ * N_ * CV * 2));
  _Float16* kTp    = (_Float16*)(ws + alloc((size_t)B_ * M_ * CQ * 2));
  _Float16* vP     = (_Float16*)(ws + alloc((size_t)B_ * CV * M_ * 2));
  _Float16* xT     = (_Float16*)(ws + alloc((size_t)B_ * N_ * C_ * 2));
  _Float16* outAT  = xT;  // xT is dead once qkv_gemm is done; reuse for outAT

  int packN = (OC * C_ + C_ * CV + OC + 255) / 256;
  pack_weights<<<packN, 256, 0, stream>>>(wq, wk, wv, wo, bq, bk, bv,
                                          wqkv_h, wo_h, biasq);
  channel_mean<<<B_ * C_, 256, 0, stream>>>(x, y0);
  se_fc1<<<B_, CV, 0, stream>>>(y0, fc1, hbuf);
  se_fc2<<<B_, C_, 0, stream>>>(hbuf, fc2, ybuf);
  transpose_x<<<dim3(N_ / 64, C_ / 32, B_), 256, 0, stream>>>(x, xT);
  qkv_gemm<<<dim3(OC / 64, N_ / 128, B_), 256, 0, stream>>>(
      xT, wqkv_h, biasq, qT, kFT, vFT);
  pool_kv<<<dim3(M_, B_), CV, 0, stream>>>(kFT, vFT, kTp, vP);
  attn_kernel<<<dim3(N_ / 64, B_), 256, 0, stream>>>(qT, kTp, vP, outAT);
  final_gemm<<<dim3(N_ / 128, C_ / 64, B_), 256, 0, stream>>>(
      outAT, wo_h, bo, x, ybuf, gamma, out);
}
